// WindowAttention_25718264168552
// MI455X (gfx1250) — matrix operand order repaired and checked by reading, not compiled, not run on hardware
//
#include <hip/hip_runtime.h>
#include <hip/hip_bf16.h>

typedef __attribute__((ext_vector_type(16))) _Float16 v16h;
typedef __attribute__((ext_vector_type(8)))  float    v8f;

#define LWIN      49
#define LPAD      64
#define DK        32
#define NHEADS    8
#define QK_STRIDE 40   // 32 + 8 pad (halves)
#define KN_STRIDE 72   // 64 + 8 pad
#define PA_STRIDE 72   // 64 + 8 pad

// A-matrix 16x32 f16 fragment (M x K), source LDS row-major [m][k].
// lanes 0-15 -> M=lane, halves 0..7 = K 0..7, halves 8..15 = K 16..23
// lanes 16-31 -> M=lane-16, halves 0..7 = K 8..15, halves 8..15 = K 24..31
__device__ __forceinline__ v16h load_a_frag(const _Float16* row0, int stride, int lane) {
  const int m   = lane & 15;
  const int khi = (lane >> 4) << 3;  // 0 or 8
  const _Float16* r = row0 + m * stride;
  v16h a;
#pragma unroll
  for (int i = 0; i < 8; ++i) a[i]     = r[khi + i];
#pragma unroll
  for (int i = 0; i < 8; ++i) a[8 + i] = r[16 + khi + i];
  return a;
}

// B-matrix 32x16 f16 fragment (K x N), source LDS n-major [n][k].
// lanes 0-15 -> N=lane, halves 0..7 = K 0..7, 8..15 = K 16..23 ; lanes 16-31 -> N=lane-16, halves 0..7 = K 8..15, 8..15 = K 24..31
__device__ __forceinline__ v16h load_b_frag(const _Float16* row0, int stride, int lane) {
  const int n   = lane & 15;
  const int khi = (lane >> 4) << 3;  // 0 or 8
  const _Float16* r = row0 + n * stride;
  v16h bf;
#pragma unroll
  for (int i = 0; i < 16; ++i) bf[i] = r[khi + (i & 7) + ((i >> 3) << 4)];
  return bf;
}

__global__ __launch_bounds__(32)
void swin_window_attn_kernel(const float* __restrict__ qg,
                             const float* __restrict__ kg,
                             const float* __restrict__ maskg,
                             const float* __restrict__ bias_table,
                             const int*   __restrict__ rel_idx,
                             float* __restrict__ score_out,
                             float* __restrict__ attn_out) {
  __shared__ _Float16 Qt[LPAD][QK_STRIDE];  // Qt[t][c] = q[c][t]  (A1)
  __shared__ _Float16 Kt[LPAD][QK_STRIDE];  // Kt[s][c] = k[c][s]  (B1, n-major)
  __shared__ _Float16 Kn[DK][KN_STRIDE];    // Kn[c][s] = k[c][s]  (B2, n-major)
  __shared__ _Float16 Pa[LPAD][PA_STRIDE];  // softmax probabilities f16 (A2)

  const int lane = threadIdx.x;             // 0..31 (wave32 == workgroup)
  const int bh   = blockIdx.x;              // 2048*8 blocks
  const int b    = bh >> 3;
  const int h    = bh & 7;

  // ---- stage q/k head slice (32 x 49 f32) into LDS as f16, coalesced b128 ----
  const size_t head_base = ((size_t)b * 256 + (size_t)h * DK) * LWIN;  // mult of 4
  const float4* q4 = (const float4*)(qg + head_base);
  const float4* k4 = (const float4*)(kg + head_base);
  for (int i4 = lane; i4 < (DK * LWIN) / 4; i4 += 32) {   // 392 float4
    float4 qv = q4[i4];
    float4 kv = k4[i4];
    float qa[4] = {qv.x, qv.y, qv.z, qv.w};
    float ka[4] = {kv.x, kv.y, kv.z, kv.w};
    const int f = i4 << 2;
#pragma unroll
    for (int e = 0; e < 4; ++e) {
      const int idx = f + e;
      const int c = (int)((unsigned)idx / 49u);
      const int t = idx - c * 49;
      Qt[t][c] = (_Float16)qa[e];
      Kt[t][c] = (_Float16)ka[e];
      Kn[c][t] = (_Float16)ka[e];
    }
  }
  // zero L padding so padded GEMM contraction terms are exact 0
#pragma unroll
  for (int t = LWIN; t < LPAD; ++t) {
    Qt[t][lane] = (_Float16)0.0f;
    Kt[t][lane] = (_Float16)0.0f;
  }
#pragma unroll
  for (int s = LWIN; s < LPAD; ++s) Kn[lane][s] = (_Float16)0.0f;
  __syncthreads();

  const float scl = 0.17677669529663687f;   // 32^-0.5
  const float* mrow_base = maskg + (size_t)(b & 63) * (LWIN * LWIN);
  float* attn_b = attn_out + (size_t)bh * (LWIN * LWIN);

  const int n     = lane & 15;              // D-fragment column
  const int mhalf = (lane >> 4) << 3;       // D-fragment row base (0 or 8)

  // ============ Phase A: S = Qt*K, in-register softmax, fill Pa ============
  for (int tt = 0; tt < 4; ++tt) {
    v16h a1 = load_a_frag(&Qt[tt * 16][0], QK_STRIDE, lane);   // K = d_k = 32 exact
    v8f d[4];
#pragma unroll
    for (int ss = 0; ss < 4; ++ss) {
      v16h b1 = load_b_frag(&Kt[ss * 16][0], QK_STRIDE, lane);
      v8f c0 = {};
      d[ss] = __builtin_amdgcn_wmma_f32_16x16x32_f16(
          false, a1, false, b1, (short)0, c0, false, false);
    }

    // softmax directly on D fragments: row t = tt*16 + mhalf + r lives in one
    // 16-lane half at VGPR r; reduce with xor-shuffles 1,2,4,8 (stays in-half).
#pragma unroll
    for (int r = 0; r < 8; ++r) {
      const int  t  = tt * 16 + mhalf + r;
      const bool tv = (t < LWIN);
      float x[4];
#pragma unroll
      for (int ss = 0; ss < 4; ++ss) {
        const int s = ss * 16 + n;
        float xv = -1e30f;
        if (tv && s < LWIN) {   // mask/bias loads: 16 consecutive lanes coalesced
          const int bi = rel_idx[t * LWIN + s];
          xv = d[ss][r] * scl + bias_table[bi * NHEADS + h] + mrow_base[t * LWIN + s];
        }
        x[ss] = xv;
      }
      float mx = fmaxf(fmaxf(x[0], x[1]), fmaxf(x[2], x[3]));
#pragma unroll
      for (int off = 1; off <= 8; off <<= 1) mx = fmaxf(mx, __shfl_xor(mx, off, 32));
      float e[4];
      float lsum = 0.0f;
#pragma unroll
      for (int ss = 0; ss < 4; ++ss) {
        const int s = ss * 16 + n;
        e[ss] = (tv && s < LWIN) ? __expf(x[ss] - mx) : 0.0f;
        lsum += e[ss];
      }
#pragma unroll
      for (int off = 1; off <= 8; off <<= 1) lsum += __shfl_xor(lsum, off, 32);
      const float inv = 1.0f / lsum;

      if (tv) {
#pragma unroll
        for (int ss = 0; ss < 4; ++ss) {
          const int s = ss * 16 + n;
          const float p = e[ss] * inv;
          Pa[t][s] = (_Float16)p;           // exact 0 for s >= 49 (padded contraction)
          if (s < LWIN) attn_b[t * LWIN + s] = p;
        }
      } else {
#pragma unroll
        for (int ss = 0; ss < 4; ++ss) Pa[t][ss * 16 + n] = (_Float16)0.0f;
      }
    }
  }
  __syncthreads();

  // ============ Phase B: score^T(t,c) = P(t,s) * Kn(c,s) ============
  float* score_b = score_out + ((size_t)b * 256 + (size_t)h * DK) * LWIN;
  for (int tt = 0; tt < 4; ++tt) {
#pragma unroll
    for (int cc = 0; cc < 2; ++cc) {
      v8f acc = {};
#pragma unroll
      for (int sc = 0; sc < 2; ++sc) {      // contraction s: 0..31, 32..63
        v16h a2 = load_a_frag(&Pa[tt * 16][sc * 32], PA_STRIDE, lane);
        v16h b2 = load_b_frag(&Kn[cc * 16][sc * 32], KN_STRIDE, lane);
        acc = __builtin_amdgcn_wmma_f32_16x16x32_f16(
            false, a2, false, b2, (short)0, acc, false, false);
      }
      const int ch = cc * 16 + n;           // channel within head
#pragma unroll
      for (int r = 0; r < 8; ++r) {
        const int t = tt * 16 + mhalf + r;
        if (t < LWIN) score_b[(size_t)ch * LWIN + t] = acc[r];
      }
    }
  }
}

extern "C" void kernel_launch(void* const* d_in, const int* in_sizes, int n_in,
                              void* d_out, int out_size, void* d_ws, size_t ws_size,
                              hipStream_t stream) {
  const float* q    = (const float*)d_in[0];
  const float* k    = (const float*)d_in[1];
  // d_in[2] == v : dead input (reference reassigns v = k)
  const float* mask = (const float*)d_in[3];
  const float* tab  = (const float*)d_in[4];
  const int*   rpi  = (const int*)d_in[5];

  float* score = (float*)d_out;                                   // 2048*256*49
  float* attn  = score + (size_t)2048 * 256 * LWIN;               // 2048*8*49*49

  dim3 grid(2048 * NHEADS);
  dim3 block(32);
  hipLaunchKernelGGL(swin_window_attn_kernel, grid, block, 0, stream,
                     q, k, mask, tab, rpi, score, attn);
}
